// Layer_23991687315654
// MI455X (gfx1250) — matrix operand order repaired and checked by reading, not compiled, not run on hardware
//
#include <hip/hip_runtime.h>

// ---------- types ----------
typedef __attribute__((ext_vector_type(16))) __bf16 v16bf;
typedef __attribute__((ext_vector_type(8)))  float  v8f;
typedef __attribute__((ext_vector_type(4)))  unsigned int u32x4;

union ABPack { u32x4 u[2]; v16bf v; };

static __device__ __forceinline__ v8f wmma_bf16(const v16bf& a, const v16bf& b, const v8f& c) {
    return __builtin_amdgcn_wmma_f32_16x16x32_bf16(false, a, false, b, (short)0, c, false, false);
}

// CDNA5 async global->LDS copy (ASYNCcnt-tracked), 16B per lane
static __device__ __forceinline__ void async_ld16(unsigned ldsoff, const __bf16* g) {
    asm volatile("global_load_async_to_lds_b128 %0, %1, off"
                 :: "v"(ldsoff), "v"((unsigned long long)(uintptr_t)g)
                 : "memory");
}
static __device__ __forceinline__ void wait_async0() {
    asm volatile("s_wait_asynccnt 0x0" ::: "memory");
}

// ---------- elementwise converts ----------
__global__ void cvt_bf16(const float* __restrict__ x, __bf16* __restrict__ y, long n) {
    long i = (long)blockIdx.x * blockDim.x + threadIdx.x;
    long stride = (long)gridDim.x * blockDim.x;
    for (; i < n; i += stride) y[i] = (__bf16)x[i];
}

// xs = bf16(x_time_new + x_freq + broadcast(x_central_new)); dims hardwired: M*D = 65536, D=256
__global__ void add3_bf16(const float* __restrict__ a, const float* __restrict__ b,
                          const float* __restrict__ c, __bf16* __restrict__ y, long n) {
    long i = (long)blockIdx.x * blockDim.x + threadIdx.x;
    long stride = (long)gridDim.x * blockDim.x;
    for (; i < n; i += stride) {
        long bt = i >> 16;
        int  k  = (int)(i & 255);
        y[i] = (__bf16)(a[i] + b[i] + c[bt * 256 + k]);
    }
}

// ---------- WMMA GEMM:  out = A @ W^T + bias (+residual) ----------
// A is 1..3 segments of (tokens,256) bf16 row-major; W is (Nout,K) bf16 row-major.
// Each wave computes a 32x64 tile: acc[2 row-tiles][4 col-tiles], B reused across row tiles.
// Segments handled by direct calls with the kernel-arg pointer so A loads lower to
// global_load_b128 (not flat_load).
__global__ __launch_bounds__(256) void wmma_gemm(
    const __bf16* __restrict__ A0, const __bf16* __restrict__ A1, const __bf16* __restrict__ A2,
    const __bf16* __restrict__ W, const float* __restrict__ bias,
    const float* __restrict__ resid, float* __restrict__ outF, __bf16* __restrict__ outB,
    int tokens, int K, int Nout)
{
    const int wave = threadIdx.x >> 5;
    const int lane = threadIdx.x & 31;
    const long row0 = ((long)blockIdx.x * 8 + wave) * 32;
    if (row0 >= tokens) return;
    const int c0 = blockIdx.y * 64;

    const int lhalf = lane >> 4;
    const int l15   = lane & 15;
    const int akb   = lhalf * 8;     // A: lane holds row M=l15, K per ISA 16-bit A layout
    const int bkb   = lhalf * 8;     // B: lane holds col N=l15, K per ISA 16-bit layout (same as A)

    const long aoff0 = (row0 + l15) * 256 + akb;
    const long aoff1 = (row0 + 16 + l15) * 256 + akb;
    const __bf16* wb = W + (long)(c0 + l15) * K + bkb;

    v8f acc[2][4] = {};

    auto runseg = [&](const __bf16* __restrict__ Aseg, const __bf16* __restrict__ wp) {
        const __bf16* a0p = Aseg + aoff0;
        const __bf16* a1p = Aseg + aoff1;
#pragma unroll
        for (int kk = 0; kk < 8; ++kk) {
            const int k = kk * 32;
            ABPack a0, a1;
            a0.u[0] = *(const u32x4*)(a0p + k);
            a0.u[1] = *(const u32x4*)(a0p + k + 16);
            a1.u[0] = *(const u32x4*)(a1p + k);
            a1.u[1] = *(const u32x4*)(a1p + k + 16);
#pragma unroll
            for (int j = 0; j < 4; ++j) {
                ABPack b;
                const __bf16* bp = wp + (long)j * 16 * K + k;
                b.u[0] = *(const u32x4*)(bp);
                b.u[1] = *(const u32x4*)(bp + 16);
                acc[0][j] = wmma_bf16(a0.v, b.v, acc[0][j]);
                acc[1][j] = wmma_bf16(a1.v, b.v, acc[1][j]);
            }
        }
    };

    runseg(A0, wb);
    if (K == 768) { runseg(A1, wb + 256); runseg(A2, wb + 512); }

#pragma unroll
    for (int mt = 0; mt < 2; ++mt) {
#pragma unroll
        for (int j = 0; j < 4; ++j) {
            const int n = c0 + 16 * j + l15;
            const float bv = bias[n];
#pragma unroll
            for (int i = 0; i < 8; ++i) {
                const long r = row0 + mt * 16 + i + lhalf * 8;
                const long idx = r * (long)Nout + n;
                const float v = acc[mt][j][i] + bv;
                if (resid) outF[idx] = v + resid[idx];
                else       outB[idx] = (__bf16)v;
            }
        }
    }
}

// ---------- persistent GRU scan ----------
// One block = 12 waves owns a 16-row hidden tile in LDS. Wh slice lives in 256 VGPRs.
// xw for step s+1 is DMA'd into LDS (global_load_async_to_lds_b128 / ASYNCcnt) while
// the WMMA phase of step s+1 runs; gate phase reads it from LDS.
// Dynamic LDS partition: hbf 8K | gh 48K | xbuf 24K = 80KB.
// token index = ((row>>Pshift) * (L<<Pshift)) + (t<<Pshift) + (row & ((1<<Pshift)-1))
__global__ __launch_bounds__(384) void gru_scan(
    const __bf16* __restrict__ xw,   // (tokens,768) bf16
    const __bf16* __restrict__ Wh,   // (768,256) bf16 row-major
    const float* __restrict__ bh,    // (768)
    __bf16* __restrict__ y,          // (tokens,256) bf16
    int Nrows, int L, int Pshift, int rev)
{
    extern __shared__ char smem_raw[];
    __bf16* hbf  = (__bf16*)(smem_raw);                  // 16*256*2 = 8192
    float*  gh   = (float*)(smem_raw + 8192);            // 16*768*4 = 49152
    __bf16* xbuf = (__bf16*)(smem_raw + 8192 + 49152);   // 16*768*2 = 24576

    const int tid  = threadIdx.x;
    const int wave = tid >> 5;
    const int lane = tid & 31;
    const int row_base = blockIdx.x * 16;

    for (int e = tid; e < 16 * 256; e += 384) hbf[e] = (__bf16)0.0f;
    __syncthreads();

    const int lhalf = lane >> 4;
    const int l15   = lane & 15;
    const int c0    = wave * 64;
    const long Lp   = (long)L << Pshift;
    const int  Pm   = (1 << Pshift) - 1;

    // preload loop-invariant Wh fragments into registers (32 tiles x 8 VGPRs)
    ABPack bReg[8][4];
    {
        const __bf16* wbase = Wh + (long)(c0 + l15) * 256 + lhalf * 8;
#pragma unroll
        for (int kk = 0; kk < 8; ++kk)
#pragma unroll
            for (int j = 0; j < 4; ++j) {
                const __bf16* bp = wbase + j * 16 * 256 + kk * 32;
                bReg[kk][j].u[0] = *(const u32x4*)(bp);
                bReg[kk][j].u[1] = *(const u32x4*)(bp + 16);
            }
    }
    float bhv[4];
#pragma unroll
    for (int j = 0; j < 4; ++j) bhv[j] = bh[c0 + 16 * j + l15];

    // per-thread async-copy chunk: 384 threads x 64B = 16 rows x 1536B (16x768 bf16)
    const int arow = tid / 24;           // 0..15
    const int aseg = tid % 24;           // 0..23 -> 64B column chunk
    const unsigned nrowA = (unsigned)(row_base + arow);
    const bool aValid = nrowA < (unsigned)Nrows;
    const long rowTokBase = aValid ? ((long)(nrowA >> Pshift) * Lp + (nrowA & Pm)) : 0;
    const unsigned ldsA = (unsigned)(uintptr_t)(xbuf + arow * 768 + aseg * 32);
    auto issue_xw = [&](int t) {
        if (aValid) {
            const __bf16* g = xw + (rowTokBase + ((long)t << Pshift)) * 768 + aseg * 32;
            async_ld16(ldsA,      g);
            async_ld16(ldsA + 16, g + 8);
            async_ld16(ldsA + 32, g + 16);
            async_ld16(ldsA + 48, g + 24);
        }
    };

    issue_xw(rev ? (L - 1) : 0);   // stage xw for step 0

    for (int s = 0; s < L; ++s) {
        const int t = rev ? (L - 1 - s) : s;

        // phase 1: gh = h @ Wh^T + bh (A from LDS, B from registers); async xw DMA in flight
        v8f acc[4] = {};
#pragma unroll
        for (int kk = 0; kk < 8; ++kk) {
            ABPack a;
            const __bf16* ap = &hbf[l15 * 256 + kk * 32 + lhalf * 8];
            a.u[0] = *(const u32x4*)(ap);
            a.u[1] = *(const u32x4*)(ap + 16);
#pragma unroll
            for (int j = 0; j < 4; ++j)
                acc[j] = wmma_bf16(a.v, bReg[kk][j].v, acc[j]);
        }
#pragma unroll
        for (int j = 0; j < 4; ++j) {
            const int n = c0 + 16 * j + l15;
#pragma unroll
            for (int i = 0; i < 8; ++i)
                gh[(i + lhalf * 8) * 768 + n] = acc[j][i] + bhv[j];
        }
        wait_async0();          // xw[t] landed in xbuf
        __syncthreads();

        // phase 2: gates + h update (all operands in LDS)
        for (int e = tid; e < 16 * 256; e += 384) {
            const int r = e >> 8, c = e & 255;
            const unsigned nrow = (unsigned)(row_base + r);
            const bool valid = nrow < (unsigned)Nrows;
            const float xr = (float)xbuf[r * 768 + c];
            const float xz = (float)xbuf[r * 768 + 256 + c];
            const float xn = (float)xbuf[r * 768 + 512 + c];
            const float gr = gh[r * 768 + c];
            const float gz = gh[r * 768 + 256 + c];
            const float gn = gh[r * 768 + 512 + c];
            const float rg = 1.0f / (1.0f + __expf(-(xr + gr)));
            const float zg = 1.0f / (1.0f + __expf(-(xz + gz)));
            const float ng = tanhf(xn + rg * gn);
            float h = (float)hbf[e];
            h = (1.0f - zg) * ng + zg * h;
            hbf[e] = (__bf16)h;
            if (valid) {
                const long tok = (long)(nrow >> Pshift) * Lp + ((long)t << Pshift) + (nrow & Pm);
                y[tok * 256 + c] = (__bf16)h;
            }
        }
        __syncthreads();
        if (s + 1 < L) issue_xw(rev ? (L - 2 - s) : (s + 1));  // overlaps next WMMA phase
    }
}

// ---------- host orchestration ----------
extern "C" void kernel_launch(void* const* d_in, const int* in_sizes, int n_in,
                              void* d_out, int out_size, void* d_ws, size_t ws_size,
                              hipStream_t stream) {
    (void)in_sizes; (void)n_in; (void)out_size; (void)ws_size;
    const int B = 8, T = 128, M = 256, D = 256;
    const long TOK = (long)B * T * M;     // 262144
    const int  NT  = B * M;               // time-GRU rows
    const int  NF  = B * T;               // freq-GRU rows

    const float* x_time    = (const float*)d_in[0];
    const float* x_central = (const float*)d_in[1];
    const float* x_freq    = (const float*)d_in[2];
    const float *Wi[5], *Wh[5], *bi[5], *bh[5];
    for (int g = 0; g < 5; ++g) {
        Wi[g] = (const float*)d_in[5 + g * 4 + 0];
        Wh[g] = (const float*)d_in[5 + g * 4 + 1];
        bi[g] = (const float*)d_in[5 + g * 4 + 2];
        bh[g] = (const float*)d_in[5 + g * 4 + 3];
    }
    const float* W_time = (const float*)d_in[25];
    const float* b_time = (const float*)d_in[26];
    const float* W_cent = (const float*)d_in[27];
    const float* b_cent = (const float*)d_in[28];
    const float* W_freq = (const float*)d_in[29];
    const float* b_freq = (const float*)d_in[30];

    float* out_time = (float*)d_out;
    float* out_cent = out_time + (long)B * T * M * D;
    float* out_freq = out_cent + (long)B * T * D;

    size_t off = 0;
    auto alloc = [&](size_t bytes) {
        void* p = (char*)d_ws + off;
        off = (off + bytes + 255) & ~(size_t)255;
        return p;
    };
    __bf16* wWi[5]; __bf16* wWh[5];
    for (int g = 0; g < 5; ++g) { wWi[g] = (__bf16*)alloc(768 * 256 * 2); wWh[g] = (__bf16*)alloc(768 * 256 * 2); }
    __bf16* wWt = (__bf16*)alloc(256 * 768 * 2);
    __bf16* wWc = (__bf16*)alloc(256 * 256 * 2);
    __bf16* wWf = (__bf16*)alloc(256 * 256 * 2);
    __bf16* xbf  = (__bf16*)alloc((size_t)TOK * 256 * 2);
    __bf16* xcbf = (__bf16*)alloc((size_t)NF * 256 * 2);
    __bf16* xw   = (__bf16*)alloc((size_t)TOK * 768 * 2);
    __bf16* y1   = (__bf16*)alloc((size_t)TOK * 256 * 2);
    __bf16* y2   = (__bf16*)alloc((size_t)TOK * 256 * 2);
    __bf16* y3   = (__bf16*)alloc((size_t)TOK * 256 * 2);
    __bf16* y4   = (__bf16*)alloc((size_t)NF * 256 * 2);
    __bf16* y5   = y1;   // reuse after x_time projection

    auto cvt = [&](const float* s, __bf16* d, long n) {
        long b = (n + 1023) / 1024; if (b > 8192) b = 8192;
        cvt_bf16<<<dim3((unsigned)b), dim3(256), 0, stream>>>(s, d, n);
    };
    auto gemm = [&](const __bf16* a0, const __bf16* a1, const __bf16* a2, const __bf16* w,
                    const float* bias, const float* res, float* of, __bf16* ob,
                    int tokens, int K, int Nout) {
        dim3 g((tokens + 255) / 256, Nout / 64);
        wmma_gemm<<<g, dim3(256), 0, stream>>>(a0, a1, a2, w, bias, res, of, ob, tokens, K, Nout);
    };
    const unsigned SCAN_LDS = 8192 + 49152 + 24576;   // 80 KB dynamic LDS
    auto scan = [&](const __bf16* xwp, const __bf16* whp, const float* bhp, __bf16* yp,
                    int N, int L, int Pshift, int rev) {
        gru_scan<<<dim3((N + 15) / 16), dim3(384), SCAN_LDS, stream>>>(xwp, whp, bhp, yp, N, L, Pshift, rev);
    };

    for (int g = 0; g < 5; ++g) { cvt(Wi[g], wWi[g], 768 * 256); cvt(Wh[g], wWh[g], 768 * 256); }
    cvt(W_time, wWt, 256 * 768);
    cvt(W_cent, wWc, 256 * 256);
    cvt(W_freq, wWf, 256 * 256);
    cvt(x_time, xbf, TOK * 256);
    cvt(x_central, xcbf, (long)NF * 256);

    // GRU t (time axis): rows = b*M+m, token order (b,t,m) -> Pshift=8, L=T
    gemm(xbf, xbf, xbf, wWi[0], bi[0], nullptr, nullptr, xw, (int)TOK, 256, 768);
    scan(xw, wWh[0], bh[0], y1, NT, T, 8, 0);
    // GRU ff (freq fwd): rows = b*T+t, L=M, Pshift=0
    gemm(xbf, xbf, xbf, wWi[1], bi[1], nullptr, nullptr, xw, (int)TOK, 256, 768);
    scan(xw, wWh[1], bh[1], y2, NF, M, 0, 0);
    // GRU fb (freq bwd)
    gemm(xbf, xbf, xbf, wWi[2], bi[2], nullptr, nullptr, xw, (int)TOK, 256, 768);
    scan(xw, wWh[2], bh[2], y3, NF, M, 0, 1);
    // central GRU: rows = b (8), L=T
    gemm(xcbf, xcbf, xcbf, wWi[3], bi[3], nullptr, nullptr, xw, NF, 256, 768);
    scan(xw, wWh[3], bh[3], y4, B, T, 0, 0);

    // x_time = x_time + [y1|y2|y3] @ W_time^T + b_time
    gemm(y1, y2, y3, wWt, b_time, x_time, out_time, nullptr, (int)TOK, 768, 256);
    // x_central = x_central + y4 @ W_cent^T + b_cent
    gemm(y4, y4, y4, wWc, b_cent, x_central, out_cent, nullptr, NF, 256, 256);

    // xs = bf16(x_time_new + x_freq + broadcast(x_central_new))
    add3_bf16<<<dim3(8192), dim3(256), 0, stream>>>(out_time, x_freq, out_cent, xbf, TOK * 256);

    // final GRU f over M
    gemm(xbf, xbf, xbf, wWi[4], bi[4], nullptr, nullptr, xw, (int)TOK, 256, 768);
    scan(xw, wWh[4], bh[4], y5, NF, M, 0, 0);
    // x_freq = x_freq + y5 @ W_freq^T + b_freq
    gemm(y5, y5, y5, wWf, b_freq, x_freq, out_freq, nullptr, (int)TOK, 256, 256);
}
